// VQ_VAE_712964571136
// MI455X (gfx1250) — matrix operand order repaired and checked by reading, not compiled, not run on hardware
//
#include <hip/hip_runtime.h>
#include <hip/hip_bf16.h>
#include <cstdint>

typedef __attribute__((ext_vector_type(16))) __bf16 v16bf;
typedef __attribute__((ext_vector_type(8)))  float  v8f;
typedef __attribute__((ext_vector_type(4)))  unsigned int u32x4;
typedef __attribute__((ext_vector_type(8)))  int    i32x8;
typedef __attribute__((ext_vector_type(4)))  int    i32x4;

#define DECAY 0.9f
#define EPSV  1e-5f

constexpr int N_ROWS = 262144;
constexpr int DIM    = 492;
constexpr int DPAD   = 512;     // padded reduction dim (16 k-steps of 32)
constexpr int KCODES = 40;
constexpr int KPAD   = 48;      // 3 column tiles of 16
constexpr int KT     = 3;
constexpr int BLOCK  = 256;     // 8 wave32
constexpr int WAVES  = BLOCK / 32;
constexpr int GRID   = 512;
constexpr int ROWS_PER_BLOCK = N_ROWS / GRID;                  // 512
constexpr int TILES_PER_WAVE = ROWS_PER_BLOCK / (WAVES * 16);  // 4

// -------- workspace layout (bytes) --------
constexpr size_t WS_EBF = 0;                                          // KPAD*DPAD bf16
constexpr size_t WS_E2  = WS_EBF + (size_t)KPAD * DPAD * 2;           // KPAD f32
constexpr size_t WS_DW  = (WS_E2 + (size_t)KPAD * 4 + 255) & ~(size_t)255; // KCODES*DPAD f32
constexpr size_t WS_CNT = WS_DW + (size_t)KCODES * DPAD * 4;          // KCODES u32
constexpr size_t WS_SXX = WS_CNT + 256;                               // 1 f32

// ---------------------------------------------------------------------------
// Setup: bf16 codebook panel (zero-padded), per-code squared norms (+inf pad),
// and zero the global accumulators (ws is poisoned before timing).
// ---------------------------------------------------------------------------
__global__ __launch_bounds__(BLOCK) void vq_setup(const float* __restrict__ E,
                                                  __bf16* __restrict__ ebf,
                                                  float* __restrict__ e2,
                                                  float* __restrict__ dwg,
                                                  unsigned int* __restrict__ cnt,
                                                  float* __restrict__ sxx) {
  const int gt = blockIdx.x * BLOCK + threadIdx.x;
  const int gs = gridDim.x * BLOCK;
  for (int i = gt; i < KPAD * DPAD; i += gs) {
    const int r = i >> 9, d = i & (DPAD - 1);
    const float v = (r < KCODES && d < DIM) ? E[r * DIM + d] : 0.0f;
    ebf[i] = (__bf16)v;
  }
  for (int i = gt; i < KCODES * DPAD; i += gs) dwg[i] = 0.0f;
  if (gt < KCODES) cnt[gt] = 0u;
  if (gt == 0) *sxx = 0.0f;
  for (int k = gt; k < KPAD; k += gs) {
    if (k < KCODES) {
      float s = 0.0f;
      for (int d = 0; d < DIM; ++d) { const float v = E[k * DIM + d]; s += v * v; }
      e2[k] = s;
    } else {
      e2[k] = __builtin_inff();   // padded codes never win the argmin
    }
  }
}

// ---------------------------------------------------------------------------
// Main pass: one sweep over the 516MB input.
// ---------------------------------------------------------------------------
__global__ __launch_bounds__(BLOCK) void vq_main(const float* __restrict__ X,
                                                 const __bf16* __restrict__ ebf_g,
                                                 const float* __restrict__ e2_g,
                                                 float* __restrict__ dw_g,
                                                 unsigned int* __restrict__ cnt_g,
                                                 float* __restrict__ sxx_g) {
  __shared__ __align__(16) __bf16 ldsE[KPAD * DPAD];   // 49152 B
  __shared__ float dwLds[KCODES * DPAD];               // 81920 B
  __shared__ unsigned int cntLds[KCODES];
  __shared__ float redLds[WAVES];

  const int tid  = threadIdx.x;
  const int lane = tid & 31;
  const int wave = tid >> 5;

  // --- stage codebook panel into LDS with the Tensor Data Mover (wave 0) ---
  if (tid < 32) {
    const unsigned int lds_off = (unsigned int)(uintptr_t)&ldsE[0];
    const unsigned long long ga = (unsigned long long)(uintptr_t)ebf_g;
    u32x4 g0;
    g0.x = 1u;                                   // count=1, user descriptor
    g0.y = lds_off;                              // lds_addr
    g0.z = (unsigned int)ga;                     // global_addr[31:0]
    g0.w = (unsigned int)((ga >> 32) & 0x01FFFFFFull) | (2u << 30);  // addr[56:32] | type=2
    i32x8 g1;
    g1[0] = (int)(1u << 16);                     // data_size=1 (2 bytes/elem)
    g1[1] = (int)(((unsigned)DPAD) << 16);       // tensor_dim0[15:0] @ bits 63:48
    g1[2] = (int)(((unsigned)KPAD) << 16);       // tensor_dim1[15:0] @ bits 95:80
    g1[3] = (int)(((unsigned)DPAD) << 16);       // tile_dim0 @ bits 127:112
    g1[4] = (int)KPAD;                           // tile_dim1 @ bits 143:128
    g1[5] = (int)DPAD;                           // tensor_dim0_stride[31:0]
    g1[6] = 0; g1[7] = 0;
    i32x4 gz4 = {0, 0, 0, 0};
    i32x8 gz8 = {0, 0, 0, 0, 0, 0, 0, 0};
    __builtin_amdgcn_tensor_load_to_lds(g0, g1, gz4, gz4, gz8, 0);
    __builtin_amdgcn_s_wait_tensorcnt(0);
  }
  for (int i = tid; i < KCODES * DPAD; i += BLOCK) dwLds[i] = 0.0f;
  if (tid < KCODES) cntLds[tid] = 0u;
  __syncthreads();

  // per-lane WMMA fragment geometry (wave32)
  const int nloc  = lane & 15;         // column within a 16-code tile / row within X tile
  const int halfA = (lane >> 4) << 3;  // A-frag K offset: 0 or 8 (interleaved halves)
  const int koffB = (lane >> 4) << 4;  // unused; B frag also uses halfA (interleaved halves)

  float e2r[KT];
#pragma unroll
  for (int t = 0; t < KT; ++t) e2r[t] = e2_g[t * 16 + nloc];

  float sqa[8];
#pragma unroll
  for (int j = 0; j < 8; ++j) sqa[j] = 0.0f;

  const int waveRow0 = blockIdx.x * ROWS_PER_BLOCK + wave * (TILES_PER_WAVE * 16);

  for (int tt = 0; tt < TILES_PER_WAVE; ++tt) {
    const int rowBase = waveRow0 + tt * 16;
    const float* xrow = X + (size_t)(rowBase + nloc) * DIM;

    const v8f vzero = {0.f, 0.f, 0.f, 0.f, 0.f, 0.f, 0.f, 0.f};
    v8f acc[KT] = {vzero, vzero, vzero};

    // convert-and-multiply step, shared by main loop and tail
    auto doStep = [&](const float* f, int d0) {
      v16bf a;
#pragma unroll
      for (int j = 0; j < 16; ++j) {
        a[j] = (__bf16)f[j];
        sqa[j & 7] = fmaf(f[j], f[j], sqa[j & 7]);   // 8 independent FMA chains
      }
#pragma unroll
      for (int t = 0; t < KT; ++t) {
        const __bf16* bp = &ldsE[(t * 16 + nloc) * DPAD + d0 + halfA];
        typedef __attribute__((ext_vector_type(8))) __bf16 v8bf_;
        const v8bf_ b0 = *(const v8bf_*)bp;
        const v8bf_ b1 = *(const v8bf_*)(bp + 16);
        const v16bf b = __builtin_shufflevector(b0, b1, 0,1,2,3,4,5,6,7,8,9,10,11,12,13,14,15);
        acc[t] = __builtin_amdgcn_wmma_f32_16x16x32_bf16(
            false, a, false, b, (short)0, acc[t], false, false);
      }
    };

    // k-steps 0..14: max index d0+31 <= 479 < 492 -> branch-free float4 loads
#pragma unroll 3
    for (int step = 0; step < 15; ++step) {
      const int d0 = step * 32;
      const float4 q0 = *(const float4*)(xrow + d0 + halfA);      // row stride 1968B, 16B aligned
      const float4 q1 = *(const float4*)(xrow + d0 + halfA + 4);
      const float4 q2 = *(const float4*)(xrow + d0 + 16 + halfA);
      const float4 q3 = *(const float4*)(xrow + d0 + 16 + halfA + 4);
      float f[16] = {q0.x, q0.y, q0.z, q0.w, q1.x, q1.y, q1.z, q1.w,
                     q2.x, q2.y, q2.z, q2.w, q3.x, q3.y, q3.z, q3.w};
      doStep(f, d0);
    }
    // k-step 15 (d0=480): only 12 of 32 lanes' chunks are in bounds
    {
      const int d0 = 480;
      float f[16];
#pragma unroll
      for (int j = 0; j < 8; ++j) {
        const int c0 = d0 + halfA + j;
        const int c1 = d0 + 16 + halfA + j;
        f[j]     = (c0 < DIM) ? xrow[c0] : 0.0f;
        f[8 + j] = (c1 < DIM) ? xrow[c1] : 0.0f;
      }
      doStep(f, d0);
    }

    // dist[m][k] = ||e_k||^2 - 2*S[m][k]; argmin per row (tie -> lowest index)
    int idx8[8];
#pragma unroll
    for (int r = 0; r < 8; ++r) {
      float bv = e2r[0] - 2.0f * acc[0][r]; int bi = nloc;
      float v1 = e2r[1] - 2.0f * acc[1][r]; if (v1 < bv) { bv = v1; bi = 16 + nloc; }
      float v2 = e2r[2] - 2.0f * acc[2][r]; if (v2 < bv) { bv = v2; bi = 32 + nloc; }
#pragma unroll
      for (int s = 1; s < 16; s <<= 1) {          // butterfly within each 16-lane half
        const float ov = __shfl_xor(bv, s, 32);
        const int   oi = __shfl_xor(bi, s, 32);
        if (ov < bv || (ov == bv && oi < bi)) { bv = ov; bi = oi; }
      }
      idx8[r] = bi;                               // half0 -> row r, half1 -> row r+8
      if (nloc == 0) atomicAdd(&cntLds[bi], 1u);
    }

    // scatter fp32 rows (L0/L2-hot float4 re-read) into per-block dw accumulator
#pragma unroll 1
    for (int m = 0; m < 16; ++m) {
      const int code = __shfl(idx8[m & 7], (m & 8) ? 16 : 0, 32);
      const float4* xr4 = (const float4*)(X + (size_t)(rowBase + m) * DIM);
      float* dwrow = &dwLds[code * DPAD];
      for (int c = lane; c < DIM / 4; c += 32) {   // DIM = 4*123, no ragged tail
        const float4 v = xr4[c];
        const int d = c * 4;
        atomicAdd(dwrow + d + 0, v.x);
        atomicAdd(dwrow + d + 1, v.y);
        atomicAdd(dwrow + d + 2, v.z);
        atomicAdd(dwrow + d + 3, v.w);
      }
    }
  }

  // --- flush per-block accumulators ---
  __syncthreads();
  for (int i = tid; i < KCODES * DPAD; i += BLOCK) {
    const float v = dwLds[i];
    if (v != 0.0f) atomicAdd(&dw_g[i], v);
  }
  if (tid < KCODES) {
    const unsigned int c = cntLds[tid];
    if (c) atomicAdd(&cnt_g[tid], c);
  }
  float s = ((sqa[0] + sqa[1]) + (sqa[2] + sqa[3])) +
            ((sqa[4] + sqa[5]) + (sqa[6] + sqa[7]));
#pragma unroll
  for (int m = 16; m >= 1; m >>= 1) s += __shfl_xor(s, m, 32);
  if (lane == 0) redLds[wave] = s;
  __syncthreads();
  if (tid == 0) {
    float tot = 0.0f;
    for (int w = 0; w < WAVES; ++w) tot += redLds[w];
    atomicAdd(sxx_g, tot);
  }
}

// ---------------------------------------------------------------------------
// Finalize: EMA updates + Laplace smoothing + closed-form loss.
// loss*N*D = sum(x^2) - 2*sum_k e'_k.dw_k + sum_k counts_k*||e'_k||^2
// ---------------------------------------------------------------------------
__global__ __launch_bounds__(BLOCK) void vq_finalize(const float* __restrict__ ema_w,
                                                     const float* __restrict__ ema_cs,
                                                     const unsigned int* __restrict__ cnt,
                                                     const float* __restrict__ dwg,
                                                     const float* __restrict__ sxx,
                                                     float* __restrict__ out) {
  __shared__ float csAdj[KCODES];
  __shared__ float cf[KCODES];
  __shared__ float rd[WAVES], re[WAVES];

  if (threadIdx.x == 0) {
    float cs[KCODES];
    float n = 0.0f;
    for (int k = 0; k < KCODES; ++k) {
      cf[k] = (float)cnt[k];
      cs[k] = ema_cs[k] * DECAY + (1.0f - DECAY) * cf[k];
      n += cs[k];
    }
    for (int k = 0; k < KCODES; ++k)
      csAdj[k] = (cs[k] + EPSV) / (n + (float)KCODES * EPSV) * n;
  }
  __syncthreads();

  float dot = 0.0f, e2c = 0.0f;
  for (int i = threadIdx.x; i < KCODES * DIM; i += BLOCK) {
    const int k = i / DIM, d = i - k * DIM;
    const float dv = dwg[k * DPAD + d];
    const float w  = ema_w[i] * DECAY + (1.0f - DECAY) * dv;
    const float e  = w / csAdj[k];
    dot += e * dv;
    e2c += cf[k] * e * e;
  }
  const int lane = threadIdx.x & 31, wave = threadIdx.x >> 5;
#pragma unroll
  for (int m = 16; m >= 1; m >>= 1) {
    dot += __shfl_xor(dot, m, 32);
    e2c += __shfl_xor(e2c, m, 32);
  }
  if (lane == 0) { rd[wave] = dot; re[wave] = e2c; }
  __syncthreads();
  if (threadIdx.x == 0) {
    float td = 0.0f, te = 0.0f;
    for (int w = 0; w < WAVES; ++w) { td += rd[w]; te += re[w]; }
    out[0] = (*sxx - 2.0f * td + te) / ((float)N_ROWS * (float)DIM);
  }
}

// ---------------------------------------------------------------------------
extern "C" void kernel_launch(void* const* d_in, const int* in_sizes, int n_in,
                              void* d_out, int out_size, void* d_ws, size_t ws_size,
                              hipStream_t stream) {
  (void)in_sizes; (void)n_in; (void)out_size; (void)ws_size;
  const float* X      = (const float*)d_in[0];  // inputs           [N, D]
  const float* E      = (const float*)d_in[1];  // embedding_weight [K, D]
  const float* ema_w  = (const float*)d_in[2];  // ema_w            [K, D]
  const float* ema_cs = (const float*)d_in[3];  // ema_cluster_size [K]

  char* ws = (char*)d_ws;
  __bf16*       ebf = (__bf16*)(ws + WS_EBF);
  float*        e2  = (float*)(ws + WS_E2);
  float*        dwg = (float*)(ws + WS_DW);
  unsigned int* cnt = (unsigned int*)(ws + WS_CNT);
  float*        sxx = (float*)(ws + WS_SXX);

  vq_setup<<<48, BLOCK, 0, stream>>>(E, ebf, e2, dwg, cnt, sxx);
  vq_main<<<GRID, BLOCK, 0, stream>>>(X, ebf, e2, dwg, cnt, sxx);
  vq_finalize<<<1, BLOCK, 0, stream>>>(ema_w, ema_cs, cnt, dwg, sxx, (float*)d_out);
}
